// MultiHeadSelfAttention_65403761983822
// MI455X (gfx1250) — matrix operand order repaired and checked by reading, not compiled, not run on hardware
//
#include <hip/hip_runtime.h>
#include <hip/hip_bf16.h>

// Problem constants (reference: B=4, S=2048, D=512, H=8, HD=64)
#define BATCH 4
#define SEQ   2048
#define DIM   512
#define NH    8
#define HD    64

typedef __bf16 bf16_t;
typedef __attribute__((ext_vector_type(16))) __bf16 v16bf;
typedef __attribute__((ext_vector_type(8)))  __bf16 v8bf;
typedef __attribute__((ext_vector_type(8)))  float  v8f;

__device__ __forceinline__ unsigned short f2bf_bits(float f) {
  unsigned u = __builtin_bit_cast(unsigned, f);
  u += 0x7FFFu + ((u >> 16) & 1u);            // round-to-nearest-even
  return (unsigned short)(u >> 16);
}

__device__ __forceinline__ bf16_t f2bf(float f) {
  unsigned short h = f2bf_bits(f);
  return __builtin_bit_cast(bf16_t, h);
}

__device__ __forceinline__ v8f zero8() {
  v8f z;
#pragma unroll
  for (int i = 0; i < 8; ++i) z[i] = 0.0f;
  return z;
}

// 16-bit A-fragment (16x32, MxK): per-lane elements are one contiguous run of 16:
// K = base+{0..15}, where base = 0 (lanes 0-15) or 16 (lanes 16-31).
__device__ __forceinline__ v16bf load_a_frag_bf16(const bf16_t* p) {
  v8bf lo = *(const v8bf*)(p);
  v8bf hi = *(const v8bf*)(p + 8);
  v16bf a;
#pragma unroll
  for (int i = 0; i < 8; ++i) { a[i] = lo[i]; a[8 + i] = hi[i]; }
  return a;
}

// ---------------------------------------------------------------------------
// Pass 0: one-shot f32 -> bf16 conversion (8 elements/thread, packed stores).
// Removes all conversion VALU from the GEMM inner loops.
// ---------------------------------------------------------------------------
__global__ __launch_bounds__(256) void cvt_bf16_kernel(
    const float* __restrict__ src, bf16_t* __restrict__ dst, int n) {
  const int i = (blockIdx.x * 256 + threadIdx.x) * 8;
  if (i >= n) return;                      // n is a multiple of 8 for all users
  const float4 a = *(const float4*)(src + i);
  const float4 b = *(const float4*)(src + i + 4);
  uint4 pk;
  pk.x = (unsigned)f2bf_bits(a.x) | ((unsigned)f2bf_bits(a.y) << 16);
  pk.y = (unsigned)f2bf_bits(a.z) | ((unsigned)f2bf_bits(a.w) << 16);
  pk.z = (unsigned)f2bf_bits(b.x) | ((unsigned)f2bf_bits(b.y) << 16);
  pk.w = (unsigned)f2bf_bits(b.z) | ((unsigned)f2bf_bits(b.w) << 16);
  *(uint4*)(dst + i) = pk;
}

// ---------------------------------------------------------------------------
// Pass 1: K = x @ Wk^T  (M=B*S=8192, N=512, K=512), bf16 in, bf16 out in two
// layouts: Krow[b*S+s][d] and KT[(b*D+d)][s].
// Block = 256 threads (8 waves); each wave computes a 16(M) x 128(N) strip.
// Inner loop is pure global_load_b128 + v_wmma.
// ---------------------------------------------------------------------------
__global__ __launch_bounds__(256) void kproj_kernel(
    const bf16_t* __restrict__ xbf, const bf16_t* __restrict__ Wkbf,
    bf16_t* __restrict__ Krow, bf16_t* __restrict__ KT) {
  const int lane = threadIdx.x & 31;
  const int wave = threadIdx.x >> 5;
  const int half = lane >> 4;
  const int l16  = lane & 15;

  const int nBlkN = DIM / 128;                 // 4
  const int m0 = (blockIdx.x / nBlkN) * 128 + wave * 16;
  const int n0 = (blockIdx.x % nBlkN) * 128;
  const int cA = half * 16;                    // A-fragment K base per half
  const int cB = half * 16;                    // B-fragment K base per half

  v8f acc[8];
#pragma unroll
  for (int j = 0; j < 8; ++j) acc[j] = zero8();

  const bf16_t* arow = xbf + (size_t)(m0 + l16) * DIM;

  for (int k0 = 0; k0 < DIM; k0 += 32) {
    v16bf a = load_a_frag_bf16(arow + k0 + cA);
#pragma unroll
    for (int j = 0; j < 8; ++j) {
      // B[k][n] = Wk[n][k]; lane holds column n = n0+16j+l16, 16 contiguous k
      const v16bf bfrag =
          *(const v16bf*)(Wkbf + (size_t)(n0 + 16 * j + l16) * DIM + k0 + cB);
      acc[j] = __builtin_amdgcn_wmma_f32_16x16x32_bf16(
          false, a, false, bfrag, (short)0, acc[j], false, false);
    }
  }

  // Epilogue: C element (M = r + half*8, N = l16) per VGPR r
#pragma unroll
  for (int j = 0; j < 8; ++j) {
    const int n = n0 + 16 * j + l16;
#pragma unroll
    for (int r = 0; r < 8; ++r) {
      const int m = m0 + r + half * 8;
      const int b = m / SEQ, s = m % SEQ;
      const bf16_t v = f2bf(acc[j][r]);
      Krow[(size_t)m * DIM + n] = v;
      KT[((size_t)b * DIM + n) * SEQ + s] = v;
    }
  }
}

// ---------------------------------------------------------------------------
// Pass 2: flash attention per (b,h). One wave owns a 16-query tile, streams
// keys 32 at a time. Computes S^T = K_keys @ Q^T so each lane holds exactly
// one query column -> per-lane scalar running max/sum; accumulates
// wV^T(64hd x 16q) with P^T assembled from registers via one shfl_xor(16).
// ---------------------------------------------------------------------------
__global__ __launch_bounds__(128) void attn_kernel(
    const bf16_t* __restrict__ Krow, const bf16_t* __restrict__ KT,
    bf16_t* __restrict__ wV) {
  const int lane = threadIdx.x & 31;
  const int wave = threadIdx.x >> 5;
  const int half = lane >> 4;
  const int l16  = lane & 15;

  const int QT_PER_BH = SEQ / 16;              // 128
  const int qt  = blockIdx.x * 4 + wave;       // 0..4095
  const int b   = qt / (NH * QT_PER_BH);
  const int rem = qt % (NH * QT_PER_BH);
  const int h   = rem / QT_PER_BH;
  const int q0  = (rem % QT_PER_BH) * 16;

  const float scale = 0.044194173824159216f;   // 1/sqrt(512) (ref scales by embed dim)

  const bf16_t* Kbh = Krow + (size_t)b * SEQ * DIM + h * HD;

  // B fragments of Q^T (32hd x 16q), chunk c covers hd [32c, 32c+32)
  v16bf bq[2];
  {
    const bf16_t* qrow = Kbh + (size_t)(q0 + l16) * DIM;
#pragma unroll
    for (int c = 0; c < 2; ++c)
      bq[c] = *(const v16bf*)(qrow + 32 * c + half * 16);
  }

  v8f acc[4];
#pragma unroll
  for (int t = 0; t < 4; ++t) acc[t] = zero8();
  float mrun = -1e30f, lrun = 0.0f;

  const bf16_t* ktbase = KT + ((size_t)b * DIM + h * HD) * SEQ;

  for (int k0 = 0; k0 < SEQ; k0 += 32) {
    // Prefetch next key block's rows into cache (global_prefetch_b8)
    if (k0 + 32 < SEQ)
      __builtin_prefetch(Kbh + (size_t)(k0 + 32 + l16) * DIM, 0, 1);

    // S^T tiles: st[tt] covers keys k0+16*tt .. +16 (rows) x 16 queries (cols)
    v8f st[2];
#pragma unroll
    for (int tt = 0; tt < 2; ++tt) {
      st[tt] = zero8();
      const bf16_t* krow_l = Kbh + (size_t)(k0 + 16 * tt + l16) * DIM;
#pragma unroll
      for (int c = 0; c < 2; ++c) {
        v16bf a = load_a_frag_bf16(krow_l + 32 * c + half * 16);
        st[tt] = __builtin_amdgcn_wmma_f32_16x16x32_bf16(
            false, a, false, bq[c], (short)0, st[tt], false, false);
      }
    }

    // Online softmax over this 32-key block. Each lane's 16 values + the
    // lane^16 partner's cover all 32 keys of one query column.
    float p0[8], p1[8];
    float tmax = -1e30f;
#pragma unroll
    for (int r = 0; r < 8; ++r) {
      p0[r] = st[0][r] * scale;
      p1[r] = st[1][r] * scale;
      tmax = fmaxf(tmax, fmaxf(p0[r], p1[r]));
    }
    tmax = fmaxf(tmax, __shfl_xor(tmax, 16, 32));
    const float mnew  = fmaxf(mrun, tmax);
    const float alpha = __expf(mrun - mnew);
    float rsum = 0.0f;
#pragma unroll
    for (int r = 0; r < 8; ++r) {
      p0[r] = __expf(p0[r] - mnew);
      p1[r] = __expf(p1[r] - mnew);
      rsum += p0[r] + p1[r];
    }
    rsum += __shfl_xor(rsum, 16, 32);
    lrun = lrun * alpha + rsum;
    mrun = mnew;
#pragma unroll
    for (int t = 0; t < 4; ++t)
#pragma unroll
      for (int r = 0; r < 8; ++r) acc[t][r] *= alpha;

    // Assemble P^T B-fragment (32 keys x 16 q): lanes<16 need keys 0..15
    // (own st0 keys r, partner's st0 keys 8+r); lanes>=16 need keys 16..31
    // (partner's st1 keys 16+r, own st1 keys 24+r).
    v16bf pf;
#pragma unroll
    for (int r = 0; r < 8; ++r) {
      const float x0 = __shfl_xor(p0[r], 16, 32);
      const float x1 = __shfl_xor(p1[r], 16, 32);
      const float elo = half ? x1    : p0[r];
      const float ehi = half ? p1[r] : x0;
      pf[r]     = f2bf(elo);
      pf[8 + r] = f2bf(ehi);
    }

    // wV^T += KT_tile(16hd x 32keys) @ P^T(32keys x 16q), 4 hd tiles
#pragma unroll
    for (int t = 0; t < 4; ++t) {
      const bf16_t* row = ktbase + (size_t)(16 * t + l16) * SEQ + k0;
      v16bf a = load_a_frag_bf16(row + half * 16);
      acc[t] = __builtin_amdgcn_wmma_f32_16x16x32_bf16(
          false, a, false, pf, (short)0, acc[t], false, false);
    }
  }

  // Normalize by row sum (per-lane scalar) and store wV[b, q, h*64+hd] in bf16.
  const float invl = 1.0f / lrun;
  bf16_t* outrow = wV + ((size_t)b * SEQ + q0 + l16) * DIM + h * HD;
#pragma unroll
  for (int t = 0; t < 4; ++t) {
#pragma unroll
    for (int r = 0; r < 8; r += 2) {
      const int hd = 16 * t + half * 8 + r;     // even -> 4B-aligned pair store
      const unsigned u0 = f2bf_bits(acc[t][r] * invl);
      const unsigned u1 = f2bf_bits(acc[t][r + 1] * invl);
      *(unsigned*)(outrow + hd) = u0 | (u1 << 16);
    }
  }
}

// ---------------------------------------------------------------------------
// Pass 3: out = wV @ Wo^T (bf16 A, bf16 B, f32 out)
// ---------------------------------------------------------------------------
__global__ __launch_bounds__(256) void oproj_kernel(
    const bf16_t* __restrict__ wV, const bf16_t* __restrict__ Wobf,
    float* __restrict__ out) {
  const int lane = threadIdx.x & 31;
  const int wave = threadIdx.x >> 5;
  const int half = lane >> 4;
  const int l16  = lane & 15;

  const int nBlkN = DIM / 128;
  const int m0 = (blockIdx.x / nBlkN) * 128 + wave * 16;
  const int n0 = (blockIdx.x % nBlkN) * 128;
  const int cA = half * 16;
  const int cB = half * 16;

  v8f acc[8];
#pragma unroll
  for (int j = 0; j < 8; ++j) acc[j] = zero8();

  const bf16_t* arow = wV + (size_t)(m0 + l16) * DIM;

  for (int k0 = 0; k0 < DIM; k0 += 32) {
    v16bf a = load_a_frag_bf16(arow + k0 + cA);
#pragma unroll
    for (int j = 0; j < 8; ++j) {
      const v16bf bfrag =
          *(const v16bf*)(Wobf + (size_t)(n0 + 16 * j + l16) * DIM + k0 + cB);
      acc[j] = __builtin_amdgcn_wmma_f32_16x16x32_bf16(
          false, a, false, bfrag, (short)0, acc[j], false, false);
    }
  }

#pragma unroll
  for (int j = 0; j < 8; ++j) {
    const int n = n0 + 16 * j + l16;
#pragma unroll
    for (int r = 0; r < 8; ++r) {
      const int m = m0 + r + half * 8;
      out[(size_t)m * DIM + n] = acc[j][r];
    }
  }
}

// ---------------------------------------------------------------------------
extern "C" void kernel_launch(void* const* d_in, const int* in_sizes, int n_in,
                              void* d_out, int out_size, void* d_ws, size_t ws_size,
                              hipStream_t stream) {
  (void)in_sizes; (void)n_in; (void)out_size; (void)ws_size;
  const float* x  = (const float*)d_in[0];
  const float* Wk = (const float*)d_in[1];
  const float* Wo = (const float*)d_in[2];
  float* out = (float*)d_out;

  const size_t nTok = (size_t)BATCH * SEQ * DIM;    // 4M elements
  const size_t nW   = (size_t)DIM * DIM;            // 256K elements
  bf16_t* xbf  = (bf16_t*)d_ws;                     // 8   MB
  bf16_t* Wkbf = xbf  + nTok;                       // 0.5 MB
  bf16_t* Wobf = Wkbf + nW;                         // 0.5 MB
  bf16_t* Krow = Wobf + nW;                         // 8   MB
  bf16_t* KT   = Krow + nTok;                       // 8   MB
  bf16_t* wVb  = KT   + nTok;                       // 8   MB

  // Pass 0: one-shot bf16 conversion of all f32 inputs (bandwidth-trivial)
  cvt_bf16_kernel<<<(int)(nTok / 8 / 256), 256, 0, stream>>>(x,  xbf,  (int)nTok);
  cvt_bf16_kernel<<<(int)(nW   / 8 / 256), 256, 0, stream>>>(Wk, Wkbf, (int)nW);
  cvt_bf16_kernel<<<(int)(nW   / 8 / 256), 256, 0, stream>>>(Wo, Wobf, (int)nW);

  // Pass 1: K projection (M=8192 rows: 64 M-blocks x 4 N-strips)
  kproj_kernel<<<256, 256, 0, stream>>>(xbf, Wkbf, Krow, KT);
  // Pass 2: streaming attention, 4096 query tiles / 4 waves per block
  attn_kernel<<<1024, 128, 0, stream>>>(Krow, KT, wVb);
  // Pass 3: output projection
  oproj_kernel<<<256, 256, 0, stream>>>(wVb, Wobf, out);
}
